// CasualSelfAttention_26207890440284
// MI455X (gfx1250) — matrix operand order repaired and checked by reading, not compiled, not run on hardware
//
#include <hip/hip_runtime.h>
#include <hip/hip_bf16.h>
#include <stdint.h>

// ---------------------------------------------------------------------------
// CDNA5 (gfx1250) causal self-attention, bf16 WMMA pipeline, fp32 accumulate.
// GEMMs use 64x64 wave tiles (acc[4][4]) for 1.0 b128-load : wmma ratio.
// ---------------------------------------------------------------------------

typedef __bf16 bf16x16 __attribute__((ext_vector_type(16)));
typedef float  f32x8   __attribute__((ext_vector_type(8)));

union FragAB { uint32_t u[8]; bf16x16 v; };

// fp32 -> bf16 round-to-nearest-even (bit trick)
__device__ __forceinline__ uint16_t f2bf(float x) {
    union { float f; uint32_t u; } v; v.f = x;
    uint32_t r = v.u + 0x7FFFu + ((v.u >> 16) & 1u);
    return (uint16_t)(r >> 16);
}

// A fragment: 16x32 bf16, row-major source, lane = M row.
// ISA layout: lanes 0-15 hold K in {0-7,16-23}, lanes 16-31 hold K in {8-15,24-31};
// VGPR v covers 2 consecutive K -> one aligned dword load per VGPR.
__device__ __forceinline__ FragAB load_a16x32(const uint16_t* __restrict__ src,
                                              int ld, int lane) {
    FragAB f;
    const int m = lane & 15, half = lane >> 4;
    const uint16_t* p = src + (size_t)m * ld + half * 8;
#pragma unroll
    for (int v = 0; v < 8; ++v) {
        const int kb = ((v & 4) << 2) + ((v & 3) << 1); // {0,2,4,6,16,18,20,22}
        f.u[v] = *reinterpret_cast<const uint32_t*>(p + kb);
    }
    return f;
}

// B fragment: 32x16 bf16, row-major source (K rows x N cols), lane = N col;
// lanes 0-15 hold K in {0-7,16-23}, lanes 16-31 hold K in {8-15,24-31} (same as A).
__device__ __forceinline__ FragAB load_b32x16(const uint16_t* __restrict__ src,
                                              int ld, int lane) {
    FragAB f;
    const int n = lane & 15, half = lane >> 4;
    const uint16_t* p = src + (size_t)(half * 8) * ld + n;
#pragma unroll
    for (int v = 0; v < 8; ++v) {
        const int kb = ((v & 4) << 2) + ((v & 3) << 1);
        f.u[v] = (uint32_t)p[(size_t)kb * ld] | ((uint32_t)p[(size_t)(kb + 1) * ld] << 16);
    }
    return f;
}

__device__ __forceinline__ f32x8 wmma_bf16(const FragAB& a, const FragAB& b, f32x8 c) {
    return __builtin_amdgcn_wmma_f32_16x16x32_bf16(false, a.v, false, b.v,
                                                   (short)0, c, false, false);
}

__device__ __forceinline__ f32x8 zero8() {
    f32x8 z;
#pragma unroll
    for (int r = 0; r < 8; ++r) z[r] = 0.0f;
    return z;
}

// ---------------------------------------------------------------------------
// Stage 0: fp32 -> bf16 conversion (grid-stride)
// ---------------------------------------------------------------------------
__global__ void cvt_f32_bf16(const float* __restrict__ in,
                             uint16_t* __restrict__ out, int n) {
    int i = blockIdx.x * blockDim.x + threadIdx.x;
    const int stride = gridDim.x * blockDim.x;
    for (; i < n; i += stride) out[i] = f2bf(in[i]);
}

// ---------------------------------------------------------------------------
// Stage 1: qkv = x @ W_attn, scattered to Q[BH][T][D], Kt[BH][D][T], V[BH][T][D]
// M=4096, K=2048, N=6144.  Wave tile 64x64, block = 8 waves covering N=512.
// ---------------------------------------------------------------------------
__global__ void __launch_bounds__(256)
qkv_gemm(const uint16_t* __restrict__ X, const uint16_t* __restrict__ W,
         uint16_t* __restrict__ Q, uint16_t* __restrict__ Kt,
         uint16_t* __restrict__ V) {
    const int K = 2048, N = 6144;
    const int lane = threadIdx.x & 31, wave = threadIdx.x >> 5;
    const int m0 = blockIdx.y * 64;
    const int n0 = blockIdx.x * 512 + wave * 64;

    f32x8 acc[4][4];
#pragma unroll
    for (int i = 0; i < 4; ++i)
#pragma unroll
        for (int j = 0; j < 4; ++j) acc[i][j] = zero8();

    for (int k0 = 0; k0 < K; k0 += 32) {
        if (k0 + 32 < K) {
            __builtin_prefetch(X + (size_t)m0 * K + k0 + 32, 0, 1);
            __builtin_prefetch(W + (size_t)(k0 + 32) * N + n0, 0, 1);
        }
        FragAB a[4];
#pragma unroll
        for (int i = 0; i < 4; ++i)
            a[i] = load_a16x32(X + (size_t)(m0 + 16 * i) * K + k0, K, lane);
#pragma unroll
        for (int j = 0; j < 4; ++j) {
            FragAB b = load_b32x16(W + (size_t)k0 * N + n0 + 16 * j, N, lane);
#pragma unroll
            for (int i = 0; i < 4; ++i)
                acc[i][j] = wmma_bf16(a[i], b, acc[i][j]);
        }
    }

    const int half = lane >> 4, nn = lane & 15;
#pragma unroll
    for (int i = 0; i < 4; ++i)
#pragma unroll
        for (int j = 0; j < 4; ++j)
#pragma unroll
            for (int r = 0; r < 8; ++r) {
                const int mg = m0 + 16 * i + r + 8 * half;   // row of x (b*T + t)
                const int ng = n0 + 16 * j + nn;             // col of qkv
                const uint16_t bv = f2bf(acc[i][j][r]);
                const int t = mg & 2047, bb = mg >> 11;
                const int sect = ng >> 11, cc = ng & 2047;
                const int h = cc >> 7, d = cc & 127;
                const size_t bh = (size_t)bb * 16 + h;
                if (sect == 0)      Q[(bh * 2048 + t) * 128 + d]  = bv;
                else if (sect == 1) Kt[(bh * 128 + d) * 2048 + t] = bv;
                else                V[(bh * 2048 + t) * 128 + d]  = bv;
            }
}

// ---------------------------------------------------------------------------
// Stage 2: flash-style causal attention. 1 wave per (b, h, 16-row query tile).
// Key blocks of 32; online softmax; P round-trips LDS (C-layout -> A-layout).
// Output y written bf16 to [B*T][C] with column h*128+d.
// ---------------------------------------------------------------------------
__global__ void __launch_bounds__(256)
attn_kernel(const uint16_t* __restrict__ Q, const uint16_t* __restrict__ Kt,
            const uint16_t* __restrict__ V, uint16_t* __restrict__ Y) {
    __shared__ __align__(16) uint16_t pbuf[8][16 * 32];

    const int lane = threadIdx.x & 31, wave = threadIdx.x >> 5;
    const int w = blockIdx.x * 8 + wave;         // 0 .. 4095
    const int qt = w & 127;                      // query tile within (b,h)
    const int h  = (w >> 7) & 15;
    const int b  = w >> 11;
    const int qt0 = qt * 16;
    const size_t bh = (size_t)b * 16 + h;

    const uint16_t* Qp = Q  + bh * 2048 * 128;   // [T][D]
    const uint16_t* Kp = Kt + bh * 128 * 2048;   // [D][T]
    const uint16_t* Vp = V  + bh * 2048 * 128;   // [T][D]

    const int half = lane >> 4, nn = lane & 15;
    const float scale = 0.08838834764831845f;    // 1/sqrt(128)

    // Q fragments reused for every key block: 4 chunks of K=32 over D=128
    FragAB qf[4];
#pragma unroll
    for (int c = 0; c < 4; ++c)
        qf[c] = load_a16x32(Qp + (size_t)qt0 * 128 + 32 * c, 128, lane);

    f32x8 o[8];
#pragma unroll
    for (int dt = 0; dt < 8; ++dt) o[dt] = zero8();
    float mrun[8], lrun[8];
#pragma unroll
    for (int r = 0; r < 8; ++r) { mrun[r] = -1e30f; lrun[r] = 0.0f; }

    const int kbmax = (qt0 + 15) >> 5;           // causal upper bound
    for (int kb = 0; kb <= kbmax; ++kb) {
        const int k0 = kb * 32;

        // ---- S = Q @ K^T for 16 queries x 32 keys -------------------------
        f32x8 s0 = zero8(), s1 = zero8();
#pragma unroll
        for (int c = 0; c < 4; ++c) {
            FragAB kf0 = load_b32x16(Kp + (size_t)(32 * c) * 2048 + k0, 2048, lane);
            FragAB kf1 = load_b32x16(Kp + (size_t)(32 * c) * 2048 + k0 + 16, 2048, lane);
            s0 = wmma_bf16(qf[c], kf0, s0);
            s1 = wmma_bf16(qf[c], kf1, s1);
        }

        // ---- scale + causal mask + online softmax -------------------------
        float p0[8], p1[8];
#pragma unroll
        for (int r = 0; r < 8; ++r) {
            const int tq  = qt0 + r + 8 * half;
            const int tk0 = k0 + nn, tk1 = k0 + 16 + nn;
            float v0 = s0[r] * scale + ((tk0 > tq) ? -1e9f : 0.0f);
            float v1 = s1[r] * scale + ((tk1 > tq) ? -1e9f : 0.0f);
            float mx = fmaxf(v0, v1);
            mx = fmaxf(mx, __shfl_xor(mx, 1, 16));
            mx = fmaxf(mx, __shfl_xor(mx, 2, 16));
            mx = fmaxf(mx, __shfl_xor(mx, 4, 16));
            mx = fmaxf(mx, __shfl_xor(mx, 8, 16));
            const float mn = fmaxf(mrun[r], mx);
            const float alpha = __expf(mrun[r] - mn);
            const float e0 = __expf(v0 - mn), e1 = __expf(v1 - mn);
            float rs = e0 + e1;
            rs += __shfl_xor(rs, 1, 16);
            rs += __shfl_xor(rs, 2, 16);
            rs += __shfl_xor(rs, 4, 16);
            rs += __shfl_xor(rs, 8, 16);
            lrun[r] = lrun[r] * alpha + rs;
            mrun[r] = mn;
            p0[r] = e0; p1[r] = e1;
#pragma unroll
            for (int dt = 0; dt < 8; ++dt) o[dt][r] *= alpha;
        }

        // ---- P: C-layout -> LDS row-major 16x32 -> A fragment -------------
#pragma unroll
        for (int r = 0; r < 8; ++r) {
            const int row = r + 8 * half;
            pbuf[wave][row * 32 + nn]      = f2bf(p0[r]);
            pbuf[wave][row * 32 + 16 + nn] = f2bf(p1[r]);
        }
        FragAB pf = load_a16x32(&pbuf[wave][0], 32, lane);

        // ---- O += P @ V (8 d-tiles of 16) ---------------------------------
#pragma unroll
        for (int dt = 0; dt < 8; ++dt) {
            FragAB vf = load_b32x16(Vp + (size_t)k0 * 128 + dt * 16, 128, lane);
            o[dt] = wmma_bf16(pf, vf, o[dt]);
        }
    }

    // ---- normalize and write y (bf16, [B*T][2048], head-major columns) ----
#pragma unroll
    for (int r = 0; r < 8; ++r) {
        const float inv = 1.0f / lrun[r];
        const int t = qt0 + r + 8 * half;
        const size_t rowoff = ((size_t)b * 2048 + t) * 2048 + h * 128;
#pragma unroll
        for (int dt = 0; dt < 8; ++dt)
            Y[rowoff + dt * 16 + nn] = f2bf(o[dt][r] * inv);
    }
}

// ---------------------------------------------------------------------------
// Stage 3: out = y @ W_proj (fp32 output).  M=4096, K=2048, N=2048.
// Wave tile 64x64.
// ---------------------------------------------------------------------------
__global__ void __launch_bounds__(256)
proj_gemm(const uint16_t* __restrict__ Yb, const uint16_t* __restrict__ W,
          float* __restrict__ out) {
    const int K = 2048, N = 2048;
    const int lane = threadIdx.x & 31, wave = threadIdx.x >> 5;
    const int m0 = blockIdx.y * 64;
    const int n0 = blockIdx.x * 512 + wave * 64;

    f32x8 acc[4][4];
#pragma unroll
    for (int i = 0; i < 4; ++i)
#pragma unroll
        for (int j = 0; j < 4; ++j) acc[i][j] = zero8();

    for (int k0 = 0; k0 < K; k0 += 32) {
        if (k0 + 32 < K) {
            __builtin_prefetch(Yb + (size_t)m0 * K + k0 + 32, 0, 1);
            __builtin_prefetch(W + (size_t)(k0 + 32) * N + n0, 0, 1);
        }
        FragAB a[4];
#pragma unroll
        for (int i = 0; i < 4; ++i)
            a[i] = load_a16x32(Yb + (size_t)(m0 + 16 * i) * K + k0, K, lane);
#pragma unroll
        for (int j = 0; j < 4; ++j) {
            FragAB b = load_b32x16(W + (size_t)k0 * N + n0 + 16 * j, N, lane);
#pragma unroll
            for (int i = 0; i < 4; ++i)
                acc[i][j] = wmma_bf16(a[i], b, acc[i][j]);
        }
    }

    const int half = lane >> 4, nn = lane & 15;
#pragma unroll
    for (int i = 0; i < 4; ++i)
#pragma unroll
        for (int j = 0; j < 4; ++j)
#pragma unroll
            for (int r = 0; r < 8; ++r) {
                const int mg = m0 + 16 * i + r + 8 * half;
                const int ng = n0 + 16 * j + nn;
                out[(size_t)mg * N + ng] = acc[i][j][r];
            }
}

// ---------------------------------------------------------------------------
// Host-side launch. Inputs: x [2,2048,2048] f32, W_attn [2048,6144] f32,
// W_proj [2048,2048] f32, mask (unused; causal mask applied analytically).
// Workspace layout (bytes): xb 16MB | Wab 24MB | Wpb 8MB | Q 16MB | Kt 16MB
//                           | V 16MB | Y 16MB  = 112MB total.
// ---------------------------------------------------------------------------
extern "C" void kernel_launch(void* const* d_in, const int* in_sizes, int n_in,
                              void* d_out, int out_size, void* d_ws, size_t ws_size,
                              hipStream_t stream) {
    const float* x  = (const float*)d_in[0];
    const float* Wa = (const float*)d_in[1];
    const float* Wp = (const float*)d_in[2];
    float* out = (float*)d_out;

    char* ws = (char*)d_ws;
    const size_t MB = 1ull << 20;
    uint16_t* xb  = (uint16_t*)(ws + 0 * MB);
    uint16_t* wab = (uint16_t*)(ws + 16 * MB);
    uint16_t* wpb = (uint16_t*)(ws + 40 * MB);
    uint16_t* Qb  = (uint16_t*)(ws + 48 * MB);
    uint16_t* Ktb = (uint16_t*)(ws + 64 * MB);
    uint16_t* Vb  = (uint16_t*)(ws + 80 * MB);
    uint16_t* Yb  = (uint16_t*)(ws + 96 * MB);

    cvt_f32_bf16<<<4096, 256, 0, stream>>>(x,  xb,  4096 * 2048);
    cvt_f32_bf16<<<4096, 256, 0, stream>>>(Wa, wab, 2048 * 6144);
    cvt_f32_bf16<<<4096, 256, 0, stream>>>(Wp, wpb, 2048 * 2048);

    qkv_gemm<<<dim3(12, 64), 256, 0, stream>>>(xb, wab, Qb, Ktb, Vb);
    attn_kernel<<<512, 256, 0, stream>>>(Qb, Ktb, Vb, Yb);
    proj_gemm<<<dim3(4, 64), 256, 0, stream>>>(Yb, wpb, out);
}
